// MultiHeadAttention_39058432589958
// MI455X (gfx1250) — matrix operand order repaired and checked by reading, not compiled, not run on hardware
//
#include <hip/hip_runtime.h>
#include <hip/hip_bf16.h>
#include <stdint.h>

// ---------------- problem constants (match reference) ----------------
constexpr int Bc = 4;
constexpr int Sc = 1024;
constexpr int Dc = 1024;
constexpr int Hc = 16;
constexpr int HDc = 64;  // head dim

typedef __attribute__((ext_vector_type(16))) __bf16 v16bf;
typedef __attribute__((ext_vector_type(8)))  __bf16 v8bf;
typedef __attribute__((ext_vector_type(8)))  float  v8f;
typedef __attribute__((ext_vector_type(4)))  unsigned int u32x4;
typedef __attribute__((ext_vector_type(8)))  int i32x8;
typedef __attribute__((ext_vector_type(4)))  int i32x4;

#if __has_builtin(__builtin_amdgcn_tensor_load_to_lds)
#define HAVE_TDM 1
#else
#define HAVE_TDM 0
#endif

__device__ __forceinline__ v8f wmma_bf16(v16bf a, v16bf b, v8f c) {
    return __builtin_amdgcn_wmma_f32_16x16x32_bf16(
        /*neg_a=*/false, a, /*neg_b=*/false, b,
        /*c_mod=*/(short)0, c, /*reuse_a=*/false, /*reuse_b=*/false);
}

// Load a 16x32 bf16 A-fragment, row-major with stride `ld`.
// Layout (ISA 7.12.2, 16-bit A 16x32): lanes 0-15 hold K={0..7,16..23},
// lanes 16-31 hold K={8..15,24..31}; row M = lane%16.
__device__ __forceinline__ v16bf load_a_frag(const __bf16* base, int ld, int lane) {
    int r  = lane & 15;
    int kb = (lane < 16) ? 0 : 8;
    const __bf16* p = base + (size_t)r * ld + kb;
    v8bf lo = *(const v8bf*)p;
    v8bf hi = *(const v8bf*)(p + 16);
    v16bf a;
#pragma unroll
    for (int i = 0; i < 8; ++i) { a[i] = lo[i]; a[i + 8] = hi[i]; }
    return a;
}

// ---------- DPP butterfly reductions within each 16-lane half (no LDS) ----------
template <int CTRL>
__device__ __forceinline__ float dpp_permf(float v) {
    int x = __float_as_int(v);
    int y = __builtin_amdgcn_update_dpp(x, x, CTRL, 0xF, 0xF, true);
    return __int_as_float(y);
}
__device__ __forceinline__ float red_max16(float v) {
    v = fmaxf(v, dpp_permf<0xB1>(v));    // quad_perm(1,0,3,2)  xor1
    v = fmaxf(v, dpp_permf<0x4E>(v));    // quad_perm(2,3,0,1)  xor2
    v = fmaxf(v, dpp_permf<0x141>(v));   // row_half_mirror     pairs across 4s
    v = fmaxf(v, dpp_permf<0x140>(v));   // row_mirror          pairs across 8s
    return v;
}
__device__ __forceinline__ float red_sum16(float v) {
    v += dpp_permf<0xB1>(v);
    v += dpp_permf<0x4E>(v);
    v += dpp_permf<0x141>(v);
    v += dpp_permf<0x140>(v);
    return v;
}

// ---------------- Tensor Data Mover: 2D tile -> LDS ----------------
__device__ __forceinline__ void wait_tensorcnt0() {
#if __has_builtin(__builtin_amdgcn_s_wait_tensorcnt)
    __builtin_amdgcn_s_wait_tensorcnt(0);
#else
    asm volatile("s_wait_tensorcnt 0x0" ::: "memory");
#endif
}

#if HAVE_TDM
// D# per CDNA5 ISA: count=1, type=2, data_size=2B,
// 2D tensor [tdim1 rows x tdim0 cols], tile [tile1 rows x tile0 cols],
// row stride = stride0 elements. Groups 2/3 zero (2D tensor).
__device__ __forceinline__ void tdm_load_2d(uint32_t lds_addr, const void* gptr,
                                            uint32_t tdim0, uint32_t tdim1,
                                            uint32_t tile0, uint32_t tile1,
                                            uint32_t stride0) {
    uint64_t ga = (uint64_t)(uintptr_t)gptr;
    u32x4 g0;
    g0[0] = 1u;                                       // count=1, user mode
    g0[1] = lds_addr;                                 // LDS byte address
    g0[2] = (uint32_t)ga;                             // global addr [31:0]
    g0[3] = (uint32_t)((ga >> 32) & 0x01FFFFFFu) | (2u << 30);  // [56:32] | type=2
    i32x8 g1;
    g1[0] = (int)(1u << 16);                          // wg_mask=0, data_size=1 (2B)
    g1[1] = (int)((tdim0 & 0xFFFFu) << 16);           // tensor_dim0 lo16 @ [63:48]
    g1[2] = (int)((tdim0 >> 16) | ((tdim1 & 0xFFFFu) << 16));
    g1[3] = (int)((tdim1 >> 16) | (tile0 << 16));     // tile_dim0 @ [127:112]
    g1[4] = (int)tile1;                               // tile_dim1, tile_dim2=0
    g1[5] = (int)stride0;                             // tensor_dim0_stride lo32
    g1[6] = 0;
    g1[7] = 0;
    i32x4 gz4 = {0, 0, 0, 0};
    i32x8 gz8 = {0, 0, 0, 0, 0, 0, 0, 0};
    __builtin_amdgcn_tensor_load_to_lds(g0, g1, gz4, gz4, gz8, 0);
}
__device__ __forceinline__ uint32_t lds_addr_of(const void* p) {
    // generic (flat) pointers to LDS carry the LDS byte offset in the low 32 bits
    return (uint32_t)(uintptr_t)p;
}
#endif

// ---------------- fp32 -> bf16 conversion ----------------
__global__ void cvt_f32_bf16_kernel(const float* __restrict__ x,
                                    __bf16* __restrict__ y, int n4) {
    int i = blockIdx.x * blockDim.x + threadIdx.x;
    if (i < n4) {
        float4 f = ((const float4*)x)[i];
        __bf16* d = y + (size_t)i * 4;
        d[0] = (__bf16)f.x; d[1] = (__bf16)f.y;
        d[2] = (__bf16)f.z; d[3] = (__bf16)f.w;
    }
}

// ---------------- GEMM + bias: C[M,N] = A[M,K] * Bw[K,N] + bias ----------------
#define BM 128
#define BN 64
#define BK 32

template <bool A_IS_BF16>
__global__ __launch_bounds__(256) void gemm_bias_kernel(
    const void* __restrict__ Aptr, const __bf16* __restrict__ Bw,
    const float* __restrict__ bias, float* __restrict__ C,
    int M, int N, int K) {
    __shared__ __align__(32) __bf16 As[BM][BK];
    __shared__ __align__(32) __bf16 Bs[BK][BN];

    const int tid  = threadIdx.x;
    const int lane = tid & 31;
    const int wid  = tid >> 5;
    const int wm   = (wid & 3) * 32;
    const int wn   = (wid >> 2) * 32;
    const int m0   = blockIdx.y * BM;
    const int n0   = blockIdx.x * BN;

    const float b_lo = bias[n0 + wn + (lane & 15)];
    const float b_hi = bias[n0 + wn + 16 + (lane & 15)];
    v8f c00, c01, c10, c11;
#pragma unroll
    for (int i = 0; i < 8; ++i) { c00[i] = b_lo; c01[i] = b_hi; c10[i] = b_lo; c11[i] = b_hi; }

    for (int k0 = 0; k0 < K; k0 += BK) {
        // ---- stage A tile (convert fp32 -> bf16 if needed) ----
        if (A_IS_BF16) {
            const __bf16* A = (const __bf16*)Aptr;
#pragma unroll
            for (int i = 0; i < 2; ++i) {
                int g = tid + i * 256;
                int r = g >> 2, cc = (g & 3) * 8;
                *(uint4*)&As[r][cc] =
                    *(const uint4*)(A + (size_t)(m0 + r) * K + k0 + cc);
            }
        } else {
            const float* A = (const float*)Aptr;
#pragma unroll
            for (int i = 0; i < 4; ++i) {
                int g = tid + i * 256;
                int r = g >> 3, cc = (g & 7) * 4;
                float4 f = *(const float4*)(A + (size_t)(m0 + r) * K + k0 + cc);
                __bf16* d = &As[r][cc];
                d[0] = (__bf16)f.x; d[1] = (__bf16)f.y;
                d[2] = (__bf16)f.z; d[3] = (__bf16)f.w;
            }
        }
        // ---- stage B tile ----
        {
            int r = tid >> 3, cc = (tid & 7) * 8;
            *(uint4*)&Bs[r][cc] =
                *(const uint4*)(Bw + (size_t)(k0 + r) * N + n0 + cc);
        }
        // ---- prefetch next-k tiles into cache (global_prefetch_b8) ----
        if (k0 + BK < K) {
            if (A_IS_BF16) {
                const __bf16* A = (const __bf16*)Aptr;
                int r = tid >> 2, cc = (tid & 3) * 8;
                __builtin_prefetch(A + (size_t)(m0 + r) * K + k0 + BK + cc, 0, 3);
            } else {
                const float* A = (const float*)Aptr;
                int r = tid >> 3, cc = (tid & 7) * 4;
                __builtin_prefetch(A + (size_t)(m0 + r) * K + k0 + BK + cc, 0, 3);
            }
            int r = tid >> 3, cc = (tid & 7) * 8;
            __builtin_prefetch(Bw + (size_t)(k0 + BK + r) * N + n0 + cc, 0, 3);
        }
        __syncthreads();

        v16bf a0 = load_a_frag(&As[wm][0],      BK, lane);
        v16bf a1 = load_a_frag(&As[wm + 16][0], BK, lane);
        v16bf b0, b1;
#pragma unroll
        for (int i = 0; i < 16; ++i) {
            int kk = (i < 8 ? i : i + 8) + ((lane >> 4) << 3);
            b0[i] = Bs[kk][wn + (lane & 15)];
            b1[i] = Bs[kk][wn + 16 + (lane & 15)];
        }

        c00 = wmma_bf16(a0, b0, c00);
        c01 = wmma_bf16(a0, b1, c01);
        c10 = wmma_bf16(a1, b0, c10);
        c11 = wmma_bf16(a1, b1, c11);
        __syncthreads();
    }

    const int mbase = m0 + wm;
    const int nbase = n0 + wn;
    const int moff  = (lane >> 4) << 3;
    const int col   = nbase + (lane & 15);
#pragma unroll
    for (int r = 0; r < 8; ++r) {
        int row = mbase + r + moff;
        C[(size_t)row * N + col]             = c00[r];
        C[(size_t)row * N + col + 16]        = c01[r];
        C[(size_t)(row + 16) * N + col]      = c10[r];
        C[(size_t)(row + 16) * N + col + 16] = c11[r];
    }
}

// ---------------- column normalization (over sequence axis) ----------------
__global__ __launch_bounds__(256) void colnorm_q_kernel(
    const float* __restrict__ X, __bf16* __restrict__ qn) {
    __shared__ float red[256];
    const int j = blockIdx.x, b = blockIdx.y, tid = threadIdx.x;
    const float* col = X + (size_t)b * Sc * Dc + j;
    float s1 = 0.f, s2 = 0.f;
    for (int s = tid; s < Sc; s += 256) {
        float x = col[(size_t)s * Dc];
        s1 += x; s2 += x * x;
    }
    red[tid] = s1; __syncthreads();
    for (int o = 128; o > 0; o >>= 1) { if (tid < o) red[tid] += red[tid + o]; __syncthreads(); }
    float sum1 = red[0]; __syncthreads();
    red[tid] = s2; __syncthreads();
    for (int o = 128; o > 0; o >>= 1) { if (tid < o) red[tid] += red[tid + o]; __syncthreads(); }
    float sum2 = red[0];
    float mean = sum1 * (1.0f / Sc);
    float var  = fmaxf(sum2 * (1.0f / Sc) - mean * mean, 0.f);
    float inv  = 1.0f / (sqrtf(var) + 0.01f);
    int h = j >> 6, d = j & 63;
    __bf16* dst = qn + ((size_t)(b * Hc + h) * Sc) * HDc + d;
    for (int s = tid; s < Sc; s += 256)
        dst[(size_t)s * HDc] = (__bf16)((col[(size_t)s * Dc] - mean) * inv);
}

__global__ __launch_bounds__(256) void colnorm_kv_kernel(
    const float* __restrict__ X, __bf16* __restrict__ kT,
    __bf16* __restrict__ vn) {
    __shared__ float red[256];
    const int j = blockIdx.x, b = blockIdx.y, tid = threadIdx.x;
    const int N2 = 2 * Dc;
    const float* col = X + (size_t)b * Sc * N2 + j;
    float s1 = 0.f, s2 = 0.f;
    for (int s = tid; s < Sc; s += 256) {
        float x = col[(size_t)s * N2];
        s1 += x; s2 += x * x;
    }
    red[tid] = s1; __syncthreads();
    for (int o = 128; o > 0; o >>= 1) { if (tid < o) red[tid] += red[tid + o]; __syncthreads(); }
    float sum1 = red[0]; __syncthreads();
    red[tid] = s2; __syncthreads();
    for (int o = 128; o > 0; o >>= 1) { if (tid < o) red[tid] += red[tid + o]; __syncthreads(); }
    float sum2 = red[0];
    float mean = sum1 * (1.0f / Sc);
    float var  = fmaxf(sum2 * (1.0f / Sc) - mean * mean, 0.f);
    float inv  = 1.0f / (sqrtf(var) + 0.01f);
    if (j < Dc) {  // K head -> transposed layout [B,H,HD,S]
        int h = j >> 6, d = j & 63;
        __bf16* dst = kT + ((size_t)(b * Hc + h) * HDc + d) * Sc;
        for (int s = tid; s < Sc; s += 256)
            dst[s] = (__bf16)((col[(size_t)s * N2] - mean) * inv);
    } else {       // V head -> [B,H,S,HD]
        int jj = j - Dc;
        int h = jj >> 6, d = jj & 63;
        __bf16* dst = vn + ((size_t)(b * Hc + h) * Sc) * HDc + d;
        for (int s = tid; s < Sc; s += 256)
            dst[(size_t)s * HDc] = (__bf16)((col[(size_t)s * N2] - mean) * inv);
    }
}

// ---------------- flash attention ----------------
// grid = (S/128, B*H); block = 256 (8 waves, 16 query rows per wave).
// K/V tiles double-buffered in LDS; staged by the Tensor Data Mover (wave 0
// issues the descriptor, TENSORcnt tracks completion) overlapped with WMMA.
__global__ __launch_bounds__(256) void attn_kernel(
    const __bf16* __restrict__ qn, const __bf16* __restrict__ kT,
    const __bf16* __restrict__ vn, __bf16* __restrict__ heads) {
    __shared__ __align__(32) __bf16 KsB[2][HDc][32];   // [buf][dim][key]  2x4KB
    __shared__ __align__(32) __bf16 VsB[2][32][HDc];   // [buf][key][dim]  2x4KB
    __shared__ __align__(32) __bf16 Ps[8][16][32];     // per-wave P       8KB

    const int tid  = threadIdx.x;
    const int lane = tid & 31;
    const int wid  = tid >> 5;
    const int bh   = blockIdx.y;
    const int b    = bh >> 4, h = bh & 15;
    const int qr   = blockIdx.x * 128 + wid * 16;

    const __bf16* qbase = qn + ((size_t)bh * Sc + qr) * HDc;
    const __bf16* kbase = kT + (size_t)bh * HDc * Sc;
    const __bf16* vbase = vn + (size_t)bh * Sc * HDc;

    // Q fragments (16 rows x 64 dims = two 16x32 A frags), hoisted.
    v16bf qf[2];
    {
        int r  = lane & 15;
        int kb = (lane < 16) ? 0 : 8;
#pragma unroll
        for (int kf = 0; kf < 2; ++kf) {
            const __bf16* p = qbase + (size_t)r * HDc + kf * 32 + kb;
            v8bf lo = *(const v8bf*)p;
            v8bf hi = *(const v8bf*)(p + 16);
#pragma unroll
            for (int i = 0; i < 8; ++i) { qf[kf][i] = lo[i]; qf[kf][i + 8] = hi[i]; }
        }
    }

    v8f o[4];
    float mrow[8], lrow[8];
#pragma unroll
    for (int j = 0; j < 4; ++j)
#pragma unroll
        for (int i = 0; i < 8; ++i) o[j][i] = 0.f;
#pragma unroll
    for (int r = 0; r < 8; ++r) { mrow[r] = -1e30f; lrow[r] = 0.f; }

    const float scale = 0.125f;  // 1/sqrt(64)
    const int NT = Sc / 32;

#if HAVE_TDM
    if (wid == 0) {
        // prologue: DMA tile 0 into buffer 0
        tdm_load_2d(lds_addr_of(&KsB[0][0][0]), kbase, Sc, HDc, 32, HDc, Sc);
        tdm_load_2d(lds_addr_of(&VsB[0][0][0]), vbase, HDc, Sc, HDc, 32, HDc);
    }
#endif

    for (int it = 0; it < NT; ++it) {
        const int cur = it & 1;
        const int k0  = it * 32;

#if HAVE_TDM
        if (wid == 0) wait_tensorcnt0();   // tile `it` resident
        __syncthreads();
        if (wid == 0 && it + 1 < NT) {     // overlap: DMA next tile into other buf
            const int nk = k0 + 32;
            tdm_load_2d(lds_addr_of(&KsB[cur ^ 1][0][0]), kbase + nk,
                        Sc, HDc, 32, HDc, Sc);
            tdm_load_2d(lds_addr_of(&VsB[cur ^ 1][0][0]), vbase + (size_t)nk * HDc,
                        HDc, Sc, HDc, 32, HDc);
        }
#else
        {
            int d = tid >> 2, cc = (tid & 3) * 8;
            *(uint4*)&KsB[cur][d][cc] = *(const uint4*)(kbase + (size_t)d * Sc + k0 + cc);
            int kk = tid >> 3, c2 = (tid & 7) * 8;
            *(uint4*)&VsB[cur][kk][c2] = *(const uint4*)(vbase + (size_t)(k0 + kk) * HDc + c2);
        }
        __syncthreads();
#endif
        const __bf16 (*Ks)[32]  = KsB[cur];
        const __bf16 (*Vs)[HDc] = VsB[cur];

        // scores: S(16x32) = Q(16x64) @ K^T(64x32)
        v8f sf[2];
#pragma unroll
        for (int j = 0; j < 2; ++j) {
            v8f z;
#pragma unroll
            for (int i = 0; i < 8; ++i) z[i] = 0.f;
#pragma unroll
            for (int kf = 0; kf < 2; ++kf) {
                v16bf bk;
#pragma unroll
                for (int i = 0; i < 16; ++i)
                    bk[i] = Ks[kf * 32 + (i < 8 ? i : i + 8) + ((lane >> 4) << 3)][j * 16 + (lane & 15)];
                z = wmma_bf16(qf[kf], bk, z);
            }
#pragma unroll
            for (int i = 0; i < 8; ++i) sf[j][i] = z[i] * scale;
        }

        // online softmax — DPP butterfly reductions, no LDS traffic
        float pr0[8], pr1[8];
#pragma unroll
        for (int r = 0; r < 8; ++r) {
            float v = red_max16(fmaxf(sf[0][r], sf[1][r]));
            float mnew = fmaxf(mrow[r], v);
            float corr = __expf(mrow[r] - mnew);
            float p0 = __expf(sf[0][r] - mnew);
            float p1 = __expf(sf[1][r] - mnew);
            float rs = red_sum16(p0 + p1);
            lrow[r] = lrow[r] * corr + rs;
            mrow[r] = mnew;
#pragma unroll
            for (int j = 0; j < 4; ++j) o[j][r] *= corr;
            pr0[r] = p0; pr1[r] = p1;
        }

        // re-layout P: C-fragment -> A-fragment via per-wave LDS region
        {
            int moff = (lane >> 4) << 3;
            int nn   = lane & 15;
#pragma unroll
            for (int r = 0; r < 8; ++r) {
                int mm = r + moff;
                Ps[wid][mm][nn]      = (__bf16)pr0[r];
                Ps[wid][mm][16 + nn] = (__bf16)pr1[r];
            }
        }
        asm volatile("s_wait_dscnt 0" ::: "memory");  // same-wave LDS RAW

        v16bf pa = load_a_frag(&Ps[wid][0][0], 32, lane);

        // O(16x64) += P(16x32) @ V(32x64)
#pragma unroll
        for (int j = 0; j < 4; ++j) {
            v16bf bv;
#pragma unroll
            for (int i = 0; i < 16; ++i)
                bv[i] = Vs[(i < 8 ? i : i + 8) + ((lane >> 4) << 3)][j * 16 + (lane & 15)];
            o[j] = wmma_bf16(pa, bv, o[j]);
        }
        __syncthreads();
    }

    // epilogue: normalize by l; store heads[B,S,D] bf16 for O-projection
    const int moff = (lane >> 4) << 3;
#pragma unroll
    for (int r = 0; r < 8; ++r) {
        float inv = 1.0f / lrow[r];
        int row = qr + r + moff;
#pragma unroll
        for (int j = 0; j < 4; ++j) {
            int col = h * HDc + j * 16 + (lane & 15);
            heads[((size_t)b * Sc + row) * Dc + col] = (__bf16)(o[j][r] * inv);
        }
    }
}

// ---------------- launcher ----------------
extern "C" void kernel_launch(void* const* d_in, const int* in_sizes, int n_in,
                              void* d_out, int out_size, void* d_ws, size_t ws_size,
                              hipStream_t stream) {
    const float* query = (const float*)d_in[0];
    const float* value = (const float*)d_in[1];
    const float* wq_k  = (const float*)d_in[2];
    const float* wq_b  = (const float*)d_in[3];
    const float* wkv_k = (const float*)d_in[4];
    const float* wkv_b = (const float*)d_in[5];
    const float* wo_k  = (const float*)d_in[6];
    const float* wo_b  = (const float*)d_in[7];
    float* out = (float*)d_out;

    char* ws = (char*)d_ws;
    __bf16* wq_bf  = (__bf16*)ws; ws += (size_t)Dc * Dc * 2;
    __bf16* wkv_bf = (__bf16*)ws; ws += (size_t)Dc * 2 * Dc * 2;
    __bf16* wo_bf  = (__bf16*)ws; ws += (size_t)Dc * Dc * 2;
    float*  out_q  = (float*)ws;  ws += (size_t)Bc * Sc * Dc * 4;
    float*  out_kv = (float*)ws;  ws += (size_t)Bc * Sc * 2 * Dc * 4;
    __bf16* qn     = (__bf16*)ws; ws += (size_t)Bc * Sc * Dc * 2;
    __bf16* kTm    = (__bf16*)ws; ws += (size_t)Bc * Sc * Dc * 2;
    __bf16* vn     = (__bf16*)ws; ws += (size_t)Bc * Sc * Dc * 2;
    __bf16* headsb = (__bf16*)ws; ws += (size_t)Bc * Sc * Dc * 2;

    const int M = Bc * Sc;  // 4096

    // 1) weight conversion fp32 -> bf16
    {
        int n4 = Dc * Dc / 4;
        cvt_f32_bf16_kernel<<<(n4 + 255) / 256, 256, 0, stream>>>(wq_k, wq_bf, n4);
        int n4kv = Dc * 2 * Dc / 4;
        cvt_f32_bf16_kernel<<<(n4kv + 255) / 256, 256, 0, stream>>>(wkv_k, wkv_bf, n4kv);
        cvt_f32_bf16_kernel<<<(n4 + 255) / 256, 256, 0, stream>>>(wo_k, wo_bf, n4);
    }

    // 2) projections
    gemm_bias_kernel<false><<<dim3(Dc / BN, M / BM), 256, 0, stream>>>(
        query, wq_bf, wq_b, out_q, M, Dc, Dc);
    gemm_bias_kernel<false><<<dim3(2 * Dc / BN, M / BM), 256, 0, stream>>>(
        value, wkv_bf, wkv_b, out_kv, M, 2 * Dc, Dc);

    // 3) column normalization + head-layout scatter
    colnorm_q_kernel<<<dim3(Dc, Bc), 256, 0, stream>>>(out_q, qn);
    colnorm_kv_kernel<<<dim3(2 * Dc, Bc), 256, 0, stream>>>(out_kv, kTm, vn);

    // 4) flash attention (TDM-staged K/V tiles, double buffered)
    attn_kernel<<<dim3(Sc / 128, Bc * Hc), 256, 0, stream>>>(qn, kTm, vn, headsb);

    // 5) output projection
    gemm_bias_kernel<true><<<dim3(Dc / BN, M / BM), 256, 0, stream>>>(
        headsb, wo_bf, wo_b, out, M, Dc, Dc);
}
